// TorchDecoder_37958920962325
// MI455X (gfx1250) — compile-verified, matrix operand order checked by reading, not run on hardware
//
#include <hip/hip_runtime.h>
#include <hip/hip_bf16.h>

// ---------------------------------------------------------------------------
// LAS decoder (3-layer LSTM + attention) for MI455X (gfx1250).
// Every matmul-like op (gate GEMMs, q/proj/vocab GEMMs, keys precompute,
// attention scores, attention context) runs through V_WMMA_F32_16X16X32_BF16.
// Working set (~80 MB) is L2-resident; per-step kernels chain on one stream.
// ---------------------------------------------------------------------------

#define NB   16
#define TI   1024
#define TSTEPS 257          // To + 1
#define DH   512            // H == D_ENC == A == 512
#define VV   8000

typedef __attribute__((ext_vector_type(16))) __bf16 v16bf;
typedef __attribute__((ext_vector_type(8)))  float  v8f;

union Frag { uint4 u[2]; v16bf v; };

// Load a 16x32 bf16 fragment (A-style layout, CDNA5 ISA §7.12.2):
// lanes 0-15: row=l, K = k0+0..7 (v0-3), k0+16..23 (v4-7)
// lanes 16-31: row=l, K = k0+8..15 (v0-3), k0+24..31 (v4-7)
// ld == 0 broadcasts one row to all 16 rows (used for vector operands).
__device__ __forceinline__ v16bf load_frag(const __hip_bfloat16* __restrict__ base,
                                           int ld, int row0, int k0, int lane)
{
  int half = (lane >> 4) & 1;
  int l    = lane & 15;
  const char* p = (const char*)(base + (size_t)(row0 + l) * ld + k0 + (half << 3));
  Frag f;
  f.u[0] = *(const uint4*)p;
  f.u[1] = *(const uint4*)(p + 32);   // +16 elements
  return f.v;
}

// Generic M=16 GEMM: C(16xN) = A(16xK) * W(NxK)^T  (+bias) (+relu) (f32/bf16)
// One wave per 16-wide column tile; f32 accumulation via WMMA bf16.
__global__ void gemm16_wmma(const __hip_bfloat16* __restrict__ A, int lda,
                            const __hip_bfloat16* __restrict__ W, int ldw,
                            float* __restrict__ Cf,
                            __hip_bfloat16* __restrict__ Cbf,
                            size_t ldc, const float* __restrict__ bias,
                            int K, int Ntiles, int do_relu)
{
  int wave = blockIdx.x * (blockDim.x >> 5) + (threadIdx.x >> 5);
  if (wave >= Ntiles) return;               // wave-uniform: EXEC stays all-ones
  int lane = threadIdx.x & 31;
  int n0   = wave << 4;
  v8f acc = {};
  for (int k0 = 0; k0 < K; k0 += 32) {
    v16bf a = load_frag(A, lda, 0,  k0, lane);
    v16bf b = load_frag(W, ldw, n0, k0, lane);
    acc = __builtin_amdgcn_wmma_f32_16x16x32_bf16(false, a, false, b,
                                                  (short)0, acc, false, false);
  }
  int half = (lane >> 4) & 1;
  int l    = lane & 15;
  int col  = n0 + l;
  float bv = bias ? bias[col] : 0.0f;
#pragma unroll
  for (int r = 0; r < 8; ++r) {
    int   m = half ? (r + 8) : r;           // D layout: vgpr r -> M=r / r+8
    float v = acc[r] + bv;
    if (do_relu) v = v > 0.0f ? v : 0.0f;
    if (Cbf) Cbf[(size_t)m * ldc + col] = __float2bfloat16(v);
    else     Cf [(size_t)m * ldc + col] = v;
  }
}

// keys[n][t][a] = (enc[n] @ Wk)[t][a], row-major bf16 (A operand of score GEMM).
// M = NB*TI rows, N = A cols, K = D.
__global__ void keys_gemm_wmma(const __hip_bfloat16* __restrict__ enc_bf,
                               const __hip_bfloat16* __restrict__ WkT,
                               __hip_bfloat16* __restrict__ keys)
{
  int wave = blockIdx.x * (blockDim.x >> 5) + (threadIdx.x >> 5);  // 32768
  int mt = wave >> 5;                       // 1024 m-tiles
  int nt = wave & 31;                       // 32 n-tiles
  int m0 = mt << 4, n0 = nt << 4;
  int lane = threadIdx.x & 31;
  v8f acc = {};
  for (int k0 = 0; k0 < DH; k0 += 32) {
    v16bf a = load_frag(enc_bf, DH, m0, k0, lane);
    v16bf b = load_frag(WkT,   DH, n0, k0, lane);
    acc = __builtin_amdgcn_wmma_f32_16x16x32_bf16(false, a, false, b,
                                                  (short)0, acc, false, false);
  }
  int half = (lane >> 4) & 1, l = lane & 15;
  int col = n0 + l;
#pragma unroll
  for (int r = 0; r < 8; ++r) {
    int m = half ? (r + 8) : r;
    keys[(size_t)(m0 + m) * DH + col] = __float2bfloat16(acc[r]);
  }
}

// score[n][t] = scale * (keys[n] @ q[n])[t]   via WMMA: A = keys tile (16 t's),
// B = q broadcast to all 16 columns (ldw = 0). 1024 waves, 16 k-steps.
__global__ void score_gemm_wmma(const __hip_bfloat16* __restrict__ keys,
                                const __hip_bfloat16* __restrict__ q_bf,
                                float* __restrict__ score)
{
  int wave = blockIdx.x * (blockDim.x >> 5) + (threadIdx.x >> 5);  // 1024
  int lane = threadIdx.x & 31;
  int g0 = wave << 4;                       // global (n,t) row
  int n  = g0 >> 10;
  const __hip_bfloat16* qn = q_bf + (size_t)n * DH;
  v8f acc = {};
  for (int k0 = 0; k0 < DH; k0 += 32) {
    v16bf a = load_frag(keys, DH, g0, k0, lane);
    v16bf b = load_frag(qn, 0, 0, k0, lane);     // broadcast q -> 16 identical cols
    acc = __builtin_amdgcn_wmma_f32_16x16x32_bf16(false, a, false, b,
                                                  (short)0, acc, false, false);
  }
  int half = (lane >> 4) & 1, l = lane & 15;
  if (l == 0) {                              // every column equals the score
    int tb = (g0 & 1023) + (half ? 8 : 0);
    float* sp = score + (size_t)n * TI;
#pragma unroll
    for (int r = 0; r < 8; ++r)
      sp[tb + r] = acc[r] * 0.04419417382415922f;   // 1/sqrt(512)
  }
}

// ctx[n][d] = (ali[n] @ enc[n])[d] via WMMA: A = ali broadcast (lda = 0),
// B = encT[n][d][t] rows. Row m=0 of D is the result (lanes 0-15, acc[0]).
__global__ void ctx_gemm_wmma(const __hip_bfloat16* __restrict__ ali_bf,
                              const __hip_bfloat16* __restrict__ encT,
                              __hip_bfloat16* __restrict__ ctx_bf,
                              __hip_bfloat16* __restrict__ xp)
{
  int wave = blockIdx.x * (blockDim.x >> 5) + (threadIdx.x >> 5);  // 512
  int lane = threadIdx.x & 31;
  int n  = wave >> 5;                       // 16 batches
  int n0 = (wave & 31) << 4;                // 32 d-tiles
  const __hip_bfloat16* an = ali_bf + (size_t)n * TI;
  const __hip_bfloat16* bn = encT + (size_t)n * DH * TI;
  v8f acc = {};
  for (int k0 = 0; k0 < TI; k0 += 32) {
    v16bf a = load_frag(an, 0, 0, k0, lane);     // broadcast ali -> 16 rows
    v16bf b = load_frag(bn, TI, n0, k0, lane);
    acc = __builtin_amdgcn_wmma_f32_16x16x32_bf16(false, a, false, b,
                                                  (short)0, acc, false, false);
  }
  int half = (lane >> 4) & 1, l = lane & 15;
  if (!half) {                              // m=0 lives in acc[0], lanes 0-15
    int d = n0 + l;
    __hip_bfloat16 v = __float2bfloat16(acc[0]);
    ctx_bf[(size_t)n * DH + d] = v;
    xp[(size_t)n * 1024 + 512 + d] = v;
  }
}

__device__ __forceinline__ float sigmoidf(float x) { return 1.0f / (1.0f + __expf(-x)); }

// Gate nonlinearity + state update; also assembles next layer's xh input:
// dst_x[:, :512] = h_new(bf16), dst_h (= xh+512) = old h of next layer.
__global__ void lstm_point(const float* __restrict__ gates,
                           const float* __restrict__ bsum,
                           float* __restrict__ c,
                           __hip_bfloat16* __restrict__ h_bf,
                           __hip_bfloat16* __restrict__ dst_x,
                           const __hip_bfloat16* __restrict__ old_h_next,
                           __hip_bfloat16* __restrict__ dst_h)
{
  int idx = blockIdx.x * blockDim.x + threadIdx.x;
  if (idx >= NB * DH) return;
  int m = idx >> 9, j = idx & 511;
  const float* g = gates + (size_t)m * 2048;
  float iv = sigmoidf(g[j]        + bsum[j]);
  float fv = sigmoidf(g[512 + j]  + bsum[512 + j]);
  float gv = tanhf   (g[1024 + j] + bsum[1024 + j]);
  float ov = sigmoidf(g[1536 + j] + bsum[1536 + j]);
  float cn = fv * c[idx] + iv * gv;
  c[idx] = cn;
  float hn = ov * tanhf(cn);
  __hip_bfloat16 hb = __float2bfloat16(hn);
  h_bf[idx] = hb;
  dst_x[(size_t)m * 1024 + j] = hb;
  if (old_h_next) dst_h[(size_t)m * 1024 + j] = old_h_next[idx];
}

// xh0 = [emb_t | ctx | h0]  (16 x 1536 bf16)
__global__ void asm0_kernel(__hip_bfloat16* __restrict__ xh0,
                            const __hip_bfloat16* __restrict__ embs,
                            const __hip_bfloat16* __restrict__ ctx_bf,
                            const __hip_bfloat16* __restrict__ h0, int t)
{
  int i = blockIdx.x * blockDim.x + threadIdx.x;
  if (i >= NB * 1536) return;
  int m = i / 1536, k = i - m * 1536;
  __hip_bfloat16 v;
  if (k < 512)       v = embs[((size_t)m * TSTEPS + t) * DH + k];
  else if (k < 1024) v = ctx_bf[m * DH + k - 512];
  else               v = h0[m * DH + k - 1024];
  xh0[i] = v;
}

__device__ __forceinline__ float wave_max(float v) {
  for (int o = 16; o > 0; o >>= 1) v = fmaxf(v, __shfl_xor(v, o, 32));
  return v;
}
__device__ __forceinline__ float wave_sum(float v) {
  for (int o = 16; o > 0; o >>= 1) v += __shfl_xor(v, o, 32);
  return v;
}

// Masked softmax over precomputed scores. One block per batch row.
// Writes ali (bf16, feeds ctx WMMA) and alis output slice (f32).
__global__ void softmax_kernel(const float* __restrict__ score,
                               const int* __restrict__ enc_len,
                               __hip_bfloat16* __restrict__ ali_bf,
                               float* __restrict__ alis_out, int t_step)
{
  __shared__ float red[8];
  __shared__ float bval;
  int n = blockIdx.x, tid = threadIdx.x;      // 256 threads
  int len = enc_len[n];
  float sc[4];
#pragma unroll
  for (int j = 0; j < 4; ++j) {
    int t = tid + j * 256;
    float s = score[(size_t)n * TI + t];
    sc[j] = (t < len) ? s : -1e30f;
  }
  int wid = tid >> 5, lane = tid & 31;
  float m = fmaxf(fmaxf(sc[0], sc[1]), fmaxf(sc[2], sc[3]));
  m = wave_max(m);
  if (lane == 0) red[wid] = m;
  __syncthreads();
  if (tid == 0) { float mm = red[0]; for (int i = 1; i < 8; ++i) mm = fmaxf(mm, red[i]); bval = mm; }
  __syncthreads();
  float bm = bval;
  float e[4], s = 0.0f;
#pragma unroll
  for (int j = 0; j < 4; ++j) { e[j] = __expf(sc[j] - bm); s += e[j]; }
  s = wave_sum(s);
  __syncthreads();
  if (lane == 0) red[wid] = s;
  __syncthreads();
  if (tid == 0) { float ss = 0.0f; for (int i = 0; i < 8; ++i) ss += red[i]; bval = ss; }
  __syncthreads();
  float inv = 1.0f / bval;
#pragma unroll
  for (int j = 0; j < 4; ++j) {
    int t = tid + j * 256;
    float v = e[j] * inv;
    ali_bf[(size_t)n * TI + t] = __float2bfloat16(v);
    alis_out[(size_t)n * TSTEPS * TI + (size_t)t_step * TI + t] = v;
  }
}

// --------------------------- precompute kernels ----------------------------
__global__ void cvt_f2bf(__hip_bfloat16* __restrict__ dst,
                         const float* __restrict__ src, long n)
{
  long i = (long)blockIdx.x * blockDim.x + threadIdx.x;
  if (i < n) dst[i] = __float2bfloat16(src[i]);
}

// dst (rows x (Ka+Kb)) bf16 = [srcA row | srcB row]; Kb may be 0.
__global__ void cvt_concat(__hip_bfloat16* __restrict__ dst,
                           const float* __restrict__ srcA, int Ka,
                           const float* __restrict__ srcB, int Kb, int rows)
{
  int K = Ka + Kb;
  long total = (long)rows * K;
  long i = (long)blockIdx.x * blockDim.x + threadIdx.x;
  if (i >= total) return;
  int r = (int)(i / K);
  int k = (int)(i - (long)r * K);
  float v = (k < Ka) ? srcA[(size_t)r * Ka + k] : srcB[(size_t)r * Kb + (k - Ka)];
  dst[i] = __float2bfloat16(v);
}

// dst (cols x rows) bf16 = transpose of src (rows x cols) f32
__global__ void tcvt(__hip_bfloat16* __restrict__ dst,
                     const float* __restrict__ src, int rows, int cols)
{
  long total = (long)rows * cols;
  long i = (long)blockIdx.x * blockDim.x + threadIdx.x;
  if (i >= total) return;
  int c = (int)(i / rows), r = (int)(i - (long)c * rows);
  dst[i] = __float2bfloat16(src[(size_t)r * cols + c]);
}

// encT[n][d][t] = enc[n][t][d] (bf16) -- B operand of the ctx WMMA.
__global__ void enc_transpose(__hip_bfloat16* __restrict__ encT,
                              const float* __restrict__ enc)
{
  long i = (long)blockIdx.x * blockDim.x + threadIdx.x;
  if (i >= (long)NB * DH * TI) return;
  int  t  = (int)(i & 1023);
  long nd = i >> 10;
  int  d  = (int)(nd & 511);
  int  n  = (int)(nd >> 9);
  encT[i] = __float2bfloat16(enc[((size_t)n * TI + t) * DH + d]);
}

__global__ void bsum_k(float* __restrict__ dst, const float* __restrict__ a,
                       const float* __restrict__ b, int n)
{
  int i = blockIdx.x * blockDim.x + threadIdx.x;
  if (i < n) dst[i] = a[i] + b[i];
}

// embs[n][t][h] = embed[token(n,t)][h], token(n,0)=sos, token(n,t)=tgt[n][t-1]
__global__ void embgather(__hip_bfloat16* __restrict__ embs,
                          const float* __restrict__ embed,
                          const int* __restrict__ tgt, const int* __restrict__ sos)
{
  int i = blockIdx.x * blockDim.x + threadIdx.x;
  if (i >= NB * TSTEPS * DH) return;
  int h  = i & 511;
  int nt = i >> 9;
  int t  = nt % TSTEPS;
  int n  = nt / TSTEPS;
  int tok = (t == 0) ? sos[0] : tgt[n * (TSTEPS - 1) + t - 1];
  embs[i] = __float2bfloat16(embed[(size_t)tok * DH + h]);
}

// ---------------------------------------------------------------------------
extern "C" void kernel_launch(void* const* d_in, const int* in_sizes, int n_in,
                              void* d_out, int out_size, void* d_ws, size_t ws_size,
                              hipStream_t stream)
{
  const float* enc_pad = (const float*)d_in[0];
  const int*   enc_len = (const int*)d_in[1];
  const int*   tgt_pad = (const int*)d_in[2];
  const int*   sos     = (const int*)d_in[3];
  const float* embed   = (const float*)d_in[4];
  const float* Wih0    = (const float*)d_in[5];
  const float* Wih12   = (const float*)d_in[6];
  const float* Whh     = (const float*)d_in[7];
  const float* bih     = (const float*)d_in[8];
  const float* bhh     = (const float*)d_in[9];
  const float* Wk      = (const float*)d_in[10];
  const float* Wq      = (const float*)d_in[11];
  const float* Wp      = (const float*)d_in[12];
  const float* bp      = (const float*)d_in[13];
  const float* Wo      = (const float*)d_in[14];
  const float* bo      = (const float*)d_in[15];

  char* base = (char*)d_ws;
  size_t off = 0;
  auto alloc = [&](size_t bytes) -> void* {
    off = (off + 255) & ~(size_t)255;
    void* p = base + off;
    off += bytes;
    return p;
  };

  __hip_bfloat16* enc_bf = (__hip_bfloat16*)alloc((size_t)NB * TI * DH * 2);
  __hip_bfloat16* encT   = (__hip_bfloat16*)alloc((size_t)NB * DH * TI * 2);
  __hip_bfloat16* keys   = (__hip_bfloat16*)alloc((size_t)NB * TI * DH * 2);
  __hip_bfloat16* embs   = (__hip_bfloat16*)alloc((size_t)NB * TSTEPS * DH * 2);
  __hip_bfloat16* W0cat  = (__hip_bfloat16*)alloc((size_t)2048 * 1536 * 2);
  __hip_bfloat16* W1cat  = (__hip_bfloat16*)alloc((size_t)2048 * 1024 * 2);
  __hip_bfloat16* W2cat  = (__hip_bfloat16*)alloc((size_t)2048 * 1024 * 2);
  __hip_bfloat16* Wp_bf  = (__hip_bfloat16*)alloc((size_t)512 * 1024 * 2);
  __hip_bfloat16* Wo_bf  = (__hip_bfloat16*)alloc((size_t)VV * DH * 2);
  __hip_bfloat16* WqT    = (__hip_bfloat16*)alloc((size_t)DH * DH * 2);
  __hip_bfloat16* WkT    = (__hip_bfloat16*)alloc((size_t)DH * DH * 2);
  float*          bsum   = (float*)alloc((size_t)3 * 2048 * 4);
  __hip_bfloat16* h_bf   = (__hip_bfloat16*)alloc((size_t)3 * NB * DH * 2);
  float*          c_st   = (float*)alloc((size_t)3 * NB * DH * 4);
  __hip_bfloat16* ctx_bf = (__hip_bfloat16*)alloc((size_t)NB * DH * 2);
  __hip_bfloat16* xh0    = (__hip_bfloat16*)alloc((size_t)NB * 1536 * 2);
  __hip_bfloat16* xh1    = (__hip_bfloat16*)alloc((size_t)NB * 1024 * 2);
  __hip_bfloat16* xh2    = (__hip_bfloat16*)alloc((size_t)NB * 1024 * 2);
  __hip_bfloat16* xp     = (__hip_bfloat16*)alloc((size_t)NB * 1024 * 2);
  float*          gates  = (float*)alloc((size_t)NB * 2048 * 4);
  __hip_bfloat16* q_bf   = (__hip_bfloat16*)alloc((size_t)NB * DH * 2);
  float*          score  = (float*)alloc((size_t)NB * TI * 4);
  __hip_bfloat16* ali_bf = (__hip_bfloat16*)alloc((size_t)NB * TI * 2);
  __hip_bfloat16* prel   = (__hip_bfloat16*)alloc((size_t)NB * DH * 2);
  (void)ws_size; (void)in_sizes; (void)n_in; (void)out_size;

  auto cdiv = [](long a, long b) { return (unsigned)((a + b - 1) / b); };
  auto gemm = [&](const __hip_bfloat16* A, int lda, const __hip_bfloat16* W, int ldw,
                  float* Cf, __hip_bfloat16* Cbf, size_t ldc,
                  const float* bias, int K, int Ntiles, int relu) {
    gemm16_wmma<<<dim3((Ntiles + 3) / 4), dim3(128), 0, stream>>>(
        A, lda, W, ldw, Cf, Cbf, ldc, bias, K, Ntiles, relu);
  };

  // ---- one-time precompute (re-run every call; deterministic) ----
  long encN = (long)NB * TI * DH;
  cvt_f2bf<<<cdiv(encN, 256), 256, 0, stream>>>(enc_bf, enc_pad, encN);
  enc_transpose<<<cdiv(encN, 256), 256, 0, stream>>>(encT, enc_pad);
  cvt_concat<<<cdiv((long)2048 * 1536, 256), 256, 0, stream>>>(
      W0cat, Wih0, 1024, Whh, 512, 2048);
  cvt_concat<<<cdiv((long)2048 * 1024, 256), 256, 0, stream>>>(
      W1cat, Wih12, 512, Whh + (size_t)2048 * 512, 512, 2048);
  cvt_concat<<<cdiv((long)2048 * 1024, 256), 256, 0, stream>>>(
      W2cat, Wih12 + (size_t)2048 * 512, 512, Whh + (size_t)2 * 2048 * 512, 512, 2048);
  cvt_concat<<<cdiv((long)512 * 1024, 256), 256, 0, stream>>>(
      Wp_bf, Wp, 1024, (const float*)nullptr, 0, 512);
  cvt_concat<<<cdiv((long)VV * DH, 256), 256, 0, stream>>>(
      Wo_bf, Wo, 512, (const float*)nullptr, 0, VV);
  tcvt<<<cdiv((long)DH * DH, 256), 256, 0, stream>>>(WqT, Wq, DH, DH);
  tcvt<<<cdiv((long)DH * DH, 256), 256, 0, stream>>>(WkT, Wk, DH, DH);
  bsum_k<<<cdiv(3 * 2048, 256), 256, 0, stream>>>(bsum, bih, bhh, 3 * 2048);
  embgather<<<cdiv((long)NB * TSTEPS * DH, 256), 256, 0, stream>>>(
      embs, embed, tgt_pad, sos);
  keys_gemm_wmma<<<dim3(8192), dim3(128), 0, stream>>>(enc_bf, WkT, keys);

  hipMemsetAsync(h_bf,   0, (size_t)3 * NB * DH * 2, stream);
  hipMemsetAsync(c_st,   0, (size_t)3 * NB * DH * 4, stream);
  hipMemsetAsync(ctx_bf, 0, (size_t)NB * DH * 2, stream);

  float* outs = (float*)d_out;
  float* alis = outs + (size_t)NB * TSTEPS * VV;

  __hip_bfloat16* h0 = h_bf;
  __hip_bfloat16* h1 = h_bf + NB * DH;
  __hip_bfloat16* h2 = h_bf + 2 * NB * DH;
  float* c0 = c_st;
  float* c1 = c_st + NB * DH;
  float* c2 = c_st + 2 * NB * DH;

  // ---- sequential decode: 257 steps ----
  for (int t = 0; t < TSTEPS; ++t) {
    asm0_kernel<<<cdiv(NB * 1536, 256), 256, 0, stream>>>(xh0, embs, ctx_bf, h0, t);

    gemm(xh0, 1536, W0cat, 1536, gates, nullptr, 2048, nullptr, 1536, 128, 0);
    lstm_point<<<32, 256, 0, stream>>>(gates, bsum, c0, h0, xh1, h1, xh1 + 512);

    gemm(xh1, 1024, W1cat, 1024, gates, nullptr, 2048, nullptr, 1024, 128, 0);
    lstm_point<<<32, 256, 0, stream>>>(gates, bsum + 2048, c1, h1, xh2, h2, xh2 + 512);

    gemm(xh2, 1024, W2cat, 1024, gates, nullptr, 2048, nullptr, 1024, 128, 0);
    lstm_point<<<32, 256, 0, stream>>>(gates, bsum + 4096, c2, h2, xp,
                                       (const __hip_bfloat16*)nullptr,
                                       (__hip_bfloat16*)nullptr);

    // attention: q (bf16) -> scores (WMMA, q broadcast) -> softmax -> ctx (WMMA)
    gemm(h2, 512, WqT, 512, nullptr, q_bf, 512, nullptr, 512, 32, 0);
    score_gemm_wmma<<<dim3(256), dim3(128), 0, stream>>>(keys, q_bf, score);
    softmax_kernel<<<NB, 256, 0, stream>>>(score, enc_len, ali_bf, alis, t);
    ctx_gemm_wmma<<<dim3(128), dim3(128), 0, stream>>>(ali_bf, encT, ctx_bf, xp);

    gemm(xp, 1024, Wp_bf, 1024, nullptr, prel, 512, bp, 1024, 32, 1);
    gemm(prel, 512, Wo_bf, 512, outs + (size_t)t * VV, nullptr,
         (size_t)TSTEPS * VV, bo, 512, 500, 0);
  }
}
